// GraphLabelPredict_70772471103910
// MI455X (gfx1250) — matrix operand order repaired and checked by reading, not compiled, not run on hardware
//
#include <hip/hip_runtime.h>
#include <hip/hip_bf16.h>

// ---------------------------------------------------------------------------
// Problem constants
// ---------------------------------------------------------------------------
#define NB      32            // batch N (becomes 32x32 "spatial" after reshape)
#define C_IN    2048
#define C_POOL  1028
#define C_MID   257
#define NCLS    10
#define HW      196           // 14*14
#define CKPAD   1056          // C_POOL padded to 33*32
#define MPAD    272           // C_MID padded to 17*16
#define HALO    34            // 32 + 1 halo each side for 3x3 conv
#define SPAT    1024          // 32*32 spatial positions

// ---------------------------------------------------------------------------
// Types for CDNA5 WMMA (wave32): v_wmma_f32_16x16x32_bf16
// ---------------------------------------------------------------------------
typedef __bf16 v16bf __attribute__((ext_vector_type(16)));   // A/B operand: 8 VGPRs
typedef float  v8f   __attribute__((ext_vector_type(8)));    // C/D operand: 8 VGPRs
typedef unsigned int uint4v __attribute__((ext_vector_type(4)));

struct bpack32 { uint4v lo, hi; };   // 32 bytes == v16bf

static __device__ __forceinline__ v16bf op_from(uint4v lo, uint4v hi) {
  bpack32 t{lo, hi};
  return __builtin_bit_cast(v16bf, t);
}

// Load a v16bf operand from two 16-byte-aligned 8-element bf16 runs.
static __device__ __forceinline__ v16bf ld_op(const unsigned short* p0,
                                              const unsigned short* p1) {
  return op_from(*(const uint4v*)p0, *(const uint4v*)p1);
}

static __device__ __forceinline__ v8f wmma_bf16(v16bf a, v16bf b, v8f c) {
  // (neg_a, A, neg_b, B, c_mod, C, reuse_a, reuse_b)
  return __builtin_amdgcn_wmma_f32_16x16x32_bf16(false, a, false, b,
                                                 (short)0, c, false, false);
}

// Pack two f32 (bit patterns) into two bf16 (truncate) with one v_perm_b32.
#define SEL_HI 0x07060302u
static __device__ __forceinline__ unsigned perm_hi(unsigned hi, unsigned lo) {
  return __builtin_amdgcn_perm(hi, lo, SEL_HI);
}

// 8 consecutive f32 -> 8 bf16 packed in one uint4 (two b128 loads + 4 v_perm).
static __device__ __forceinline__ uint4v pack8(const float* p) {
  uint4v a = *(const uint4v*)p;
  uint4v b = *(const uint4v*)(p + 4);
  uint4v r;
  r.x = perm_hi(a.y, a.x);
  r.y = perm_hi(a.w, a.z);
  r.z = perm_hi(b.y, b.x);
  r.w = perm_hi(b.w, b.z);
  return r;
}

// round-to-nearest-even f32 -> bf16 bits (for the small pack kernels)
static __device__ __forceinline__ unsigned short f2bf(float f) {
  unsigned u = __builtin_bit_cast(unsigned, f);
  unsigned r = u + 0x7FFFu + ((u >> 16) & 1u);
  return (unsigned short)(r >> 16);
}

// ---------------------------------------------------------------------------
// Stage 0a: pack conv1 weights f32 [257][1028][3][3] -> bf16 Wp [9][272][1056]
// ---------------------------------------------------------------------------
__global__ void k_pack_weights(const float* __restrict__ W,
                               unsigned short* __restrict__ Wp) {
  int t = blockIdx.x * blockDim.x + threadIdx.x;
  if (t >= 9 * MPAD * CKPAD) return;
  int p   = t / (MPAD * CKPAD);
  int rem = t - p * (MPAD * CKPAD);
  int oc  = rem / CKPAD;
  int cc  = rem - oc * CKPAD;
  float v = (oc < C_MID && cc < C_POOL) ? W[(oc * C_POOL + cc) * 9 + p] : 0.0f;
  Wp[t] = f2bf(v);
}

// ---------------------------------------------------------------------------
// Stage 0b: zero the halo'd conv input Xt [34][34][1056] bf16
// ---------------------------------------------------------------------------
__global__ void k_zero_xt(unsigned short* __restrict__ Xt) {
  int t = blockIdx.x * blockDim.x + threadIdx.x;
  if (t < HALO * HALO * CKPAD) Xt[t] = 0;
}

// ---------------------------------------------------------------------------
// Stage 1: per-channel Gram  G[c][m*32+n] = (1/196) * <F[m,c,:], F[n,c,:]>
// One wave per channel; fused f32->bf16 packing (v_perm) feeding
// v_wmma_f32_16x16x32_bf16; K=196 handled as 6 full 32-steps + peeled 4-tail.
// Exploits symmetry: row data serves both A and B operand roles.
// ---------------------------------------------------------------------------
__global__ void k_gram_wmma(const float* __restrict__ F,
                            float* __restrict__ G) {
  int wave = (blockIdx.x * blockDim.x + threadIdx.x) >> 5;   // channel, exact 2048
  int lane = threadIdx.x & 31;
  int rlo  = lane & 15;
  int grp  = lane >> 4;                                      // lane half

  const float* r0 = F + ((size_t)rlo * C_IN + (size_t)wave) * HW;   // row rlo
  const float* r1 = r0 + (size_t)16 * C_IN * HW;                    // row rlo+16

  v8f c00 = {}, c01 = {}, c10 = {}, c11 = {};

  #pragma unroll 1
  for (int k0 = 0; k0 < 192; k0 += 32) {
    int ka = k0 + grp * 8;     // A: two 8-runs {ka, ka+16}
    int kb = k0 + grp * 16;    // unused; B loads the same 8-runs as A
    v16bf a0 = op_from(pack8(r0 + ka), pack8(r0 + ka + 16));
    v16bf a1 = op_from(pack8(r1 + ka), pack8(r1 + ka + 16));
    v16bf b0 = op_from(pack8(r0 + ka), pack8(r0 + ka + 16));
    v16bf b1 = op_from(pack8(r1 + ka), pack8(r1 + ka + 16));
    c00 = wmma_bf16(a0, b0, c00);
    c01 = wmma_bf16(a0, b1, c01);
    c10 = wmma_bf16(a1, b0, c10);
    c11 = wmma_bf16(a1, b1, c11);
  }

  // Tail: k = 192..195 valid (4 of 32). Both A and B layouts place them in
  // elements 0..3 of lane-half 0; everything else zero. No divergence, no OOB.
  {
    uint4v z = {0u, 0u, 0u, 0u};
    unsigned msk = (grp == 0) ? 0xFFFFFFFFu : 0u;
    uint4v l0 = *(const uint4v*)(r0 + 192);
    uint4v l1 = *(const uint4v*)(r1 + 192);
    uint4v p0 = z, p1 = z;
    p0.x = perm_hi(l0.y, l0.x) & msk;
    p0.y = perm_hi(l0.w, l0.z) & msk;
    p1.x = perm_hi(l1.y, l1.x) & msk;
    p1.y = perm_hi(l1.w, l1.z) & msk;
    v16bf t0 = op_from(p0, z);
    v16bf t1 = op_from(p1, z);
    c00 = wmma_bf16(t0, t0, c00);
    c01 = wmma_bf16(t0, t1, c01);
    c10 = wmma_bf16(t1, t0, c10);
    c11 = wmma_bf16(t1, t1, c11);
  }

  const float s = 1.0f / 196.0f;
  float* g   = G + wave * SPAT;
  int   nn   = lane & 15;
  int   madd = grp * 8;
  #pragma unroll
  for (int r = 0; r < 8; ++r) {
    int m = r + madd;
    g[m * 32 + nn]             = c00[r] * s;
    g[m * 32 + 16 + nn]        = c01[r] * s;
    g[(m + 16) * 32 + nn]      = c10[r] * s;
    g[(m + 16) * 32 + 16 + nn] = c11[r] * s;
  }
}

// ---------------------------------------------------------------------------
// Stage 2: adaptive pool 2048->1028 (overlapping bins) + [31,31]->1.0 fixup,
// raw-reshape scatter into the halo'd, transposed, bf16 conv input
// Xt[(q+1)][(r+1)][p] where x[p][q][r] = pooled_flat[p*1024+q*32+r].
// ---------------------------------------------------------------------------
__global__ void k_pool_reshape(const float* __restrict__ G,
                               unsigned short* __restrict__ Xt) {
  int t = blockIdx.x * blockDim.x + threadIdx.x;   // exact 1028*1024
  int p = t >> 10;
  int s = t & 1023;
  int m   = t / (NB * C_POOL);
  int rem = t - m * (NB * C_POOL);
  int n   = rem / C_POOL;
  int c   = rem - n * C_POOL;

  float v;
  if (m == 31 && n == 31) {
    v = 1.0f;
  } else {
    int s0 = (c * 512) / 257;                       // floor(c*2048/1028)
    int e0 = ((c + 1) * 512 + 256) / 257;           // ceil((c+1)*2048/1028)
    float acc = 0.0f;
    for (int cc = s0; cc < e0; ++cc) acc += G[cc * SPAT + m * 32 + n];
    v = acc / (float)(e0 - s0);
  }
  int q = s >> 5, r = s & 31;
  Xt[((q + 1) * HALO + (r + 1)) * CKPAD + p] = f2bf(v);
}

// ---------------------------------------------------------------------------
// Stage 3: 3x3 conv as implicit GEMM (bf16 WMMA, f32 accum) + fused BN + ReLU.
// Block = 8 waves sharing one 16-row M tile; per 3x3 tap the 33 KB weight
// panel is staged into LDS with global_load_async_to_lds_b128 (ASYNCcnt) and
// read back with ds_load. Each wave covers 2 spatial rows (4 accum tiles), so
// steady state per K-step = 1 LDS A-load + 4 B-loads + 4 WMMA.
// ---------------------------------------------------------------------------
__global__ void k_conv3x3_wmma(const unsigned short* __restrict__ Wp,
                               const unsigned short* __restrict__ Xt,
                               const float* __restrict__ bn_gamma,
                               const float* __restrict__ bn_beta,
                               const float* __restrict__ bn_mean,
                               const float* __restrict__ bn_var,
                               float* __restrict__ Y) {
  __shared__ __align__(16) unsigned short lds_a[16 * CKPAD];   // 33 KB

  int tid  = threadIdx.x;
  int lane = tid & 31;
  int wv   = tid >> 5;                          // 0..7 wave in block
  int mt   = blockIdx.x >> 1;                   // 0..16 : M tile (shared by block)
  int qp   = (blockIdx.x & 1) * 8 + wv;         // 0..15 : spatial row pair
  int q0   = qp * 2, q1 = q0 + 1;
  int rlo  = lane & 15;
  int grp  = lane >> 4;

  v8f a00 = {}, a01 = {}, a10 = {}, a11 = {};

  const unsigned short* wtile = Wp + (size_t)(mt * 16) * CKPAD;

  #pragma unroll 1
  for (int p = 0; p < 9; ++p) {
    __syncthreads();   // WAR: previous tap's LDS reads complete before overwrite
    {
      const unsigned short* src = wtile + (size_t)p * (MPAD * CKPAD);
      #pragma unroll 1
      for (int ch = tid; ch < (16 * CKPAD) / 8; ch += 256) {
        unsigned lds_off = (unsigned)(size_t)&lds_a[ch * 8];   // low32 = LDS offset
        const unsigned short* g = src + ch * 8;
        asm volatile("global_load_async_to_lds_b128 %0, %1, off"
                     :: "v"(lds_off), "v"(g) : "memory");
      }
      asm volatile("s_wait_asynccnt 0x0" ::: "memory");
    }
    __syncthreads();

    int dh = p / 3 - 1, dw = p % 3 - 1;
    const unsigned short* xb00 =
        Xt + ((size_t)(q0 + dh + 1) * HALO + (size_t)(rlo + dw + 1)) * CKPAD;
    const unsigned short* xb01 = xb00 + 16 * CKPAD;
    const unsigned short* xb10 = xb00 + HALO * CKPAD;      // q1 = q0 + 1
    const unsigned short* xb11 = xb10 + 16 * CKPAD;

    #pragma unroll 1
    for (int k0 = 0; k0 < CKPAD; k0 += 32) {
      const unsigned short* pa = &lds_a[rlo * CKPAD + k0 + grp * 16];
      v16bf A   = ld_op(pa, pa + 8);
      int   kb  = k0 + grp * 16;
      v16bf B00 = ld_op(xb00 + kb, xb00 + kb + 8);
      v16bf B01 = ld_op(xb01 + kb, xb01 + kb + 8);
      v16bf B10 = ld_op(xb10 + kb, xb10 + kb + 8);
      v16bf B11 = ld_op(xb11 + kb, xb11 + kb + 8);
      a00 = wmma_bf16(A, B00, a00);
      a01 = wmma_bf16(A, B01, a01);
      a10 = wmma_bf16(A, B10, a10);
      a11 = wmma_bf16(A, B11, a11);
    }
  }

  // Fused BatchNorm (eval) + ReLU epilogue; drop padded oc >= 257.
  int nn    = lane & 15;
  int mbase = mt * 16 + grp * 8;
  #pragma unroll
  for (int r = 0; r < 8; ++r) {
    int oc = mbase + r;
    if (oc < C_MID) {
      float sc = bn_gamma[oc] * rsqrtf(bn_var[oc] + 1e-5f);
      float sh = bn_beta[oc] - bn_mean[oc] * sc;
      float v00 = a00[r] * sc + sh; v00 = v00 > 0.0f ? v00 : 0.0f;
      float v01 = a01[r] * sc + sh; v01 = v01 > 0.0f ? v01 : 0.0f;
      float v10 = a10[r] * sc + sh; v10 = v10 > 0.0f ? v10 : 0.0f;
      float v11 = a11[r] * sc + sh; v11 = v11 > 0.0f ? v11 : 0.0f;
      Y[oc * SPAT + q0 * 32 + nn]       = v00;
      Y[oc * SPAT + q0 * 32 + 16 + nn]  = v01;
      Y[oc * SPAT + q1 * 32 + nn]       = v10;
      Y[oc * SPAT + q1 * 32 + 16 + nn]  = v11;
    }
  }
}

// ---------------------------------------------------------------------------
// Stage 4: 1x1 conv (10 x 257) + bias. Tiny (~5 MFLOP) -> plain VALU kernel.
// ---------------------------------------------------------------------------
__global__ void k_conv1x1(const float* __restrict__ Y,
                          const float* __restrict__ W2,
                          const float* __restrict__ B2,
                          float* __restrict__ out) {
  int t = blockIdx.x * blockDim.x + threadIdx.x;   // exact 10*1024
  int oc = t >> 10;
  int s  = t & 1023;
  float acc = B2[oc];
  #pragma unroll 1
  for (int c = 0; c < C_MID; ++c)
    acc += W2[oc * C_MID + c] * Y[c * SPAT + s];
  out[t] = acc;
}

// ---------------------------------------------------------------------------
// Workspace layout (bytes, 256-aligned), total ~16.3 MiB
// ---------------------------------------------------------------------------
#define OFF_WP 0u             // 9*272*1056 bf16  = 5,170,176
#define OFF_XT 5170176u       // 34*34*1056 bf16  = 2,441,472
#define OFF_G  7611648u       // 2048*1024 f32    = 8,388,608
#define OFF_Y  16000256u      // 257*1024 f32     = 1,052,672

extern "C" void kernel_launch(void* const* d_in, const int* in_sizes, int n_in,
                              void* d_out, int out_size, void* d_ws, size_t ws_size,
                              hipStream_t stream) {
  const float* features = (const float*)d_in[0];
  const float* conv1_w  = (const float*)d_in[1];
  const float* bn_gamma = (const float*)d_in[2];
  const float* bn_beta  = (const float*)d_in[3];
  const float* bn_mean  = (const float*)d_in[4];
  const float* bn_var   = (const float*)d_in[5];
  const float* conv2_w  = (const float*)d_in[6];
  const float* conv2_b  = (const float*)d_in[7];
  float* out = (float*)d_out;

  char* ws = (char*)d_ws;
  unsigned short* Wp = (unsigned short*)(ws + OFF_WP);
  unsigned short* Xt = (unsigned short*)(ws + OFF_XT);
  float*          G  = (float*)(ws + OFF_G);
  float*          Y  = (float*)(ws + OFF_Y);

  const int TB = 256;

  // 0a: pack conv1 weights -> bf16 [tap][272][1056]
  {
    int n = 9 * MPAD * CKPAD;
    k_pack_weights<<<(n + TB - 1) / TB, TB, 0, stream>>>(conv1_w, Wp);
  }
  // 0b: zero halo'd conv input
  {
    int n = HALO * HALO * CKPAD;
    k_zero_xt<<<(n + TB - 1) / TB, TB, 0, stream>>>(Xt);
  }
  // 1: per-channel Gram via bf16 WMMA, fused f32->bf16 pack (exact 2048 waves)
  k_gram_wmma<<<(C_IN * 32) / TB, TB, 0, stream>>>(features, G);

  // 2: adaptive pool + raw reshape into halo'd bf16 conv input (exact grid)
  k_pool_reshape<<<(C_POOL * SPAT) / TB, TB, 0, stream>>>(G, Xt);

  // 3: 3x3 conv implicit GEMM + BN + ReLU (34 blocks x 8 waves, LDS-staged A)
  k_conv3x3_wmma<<<34, TB, 0, stream>>>(
      Wp, Xt, bn_gamma, bn_beta, bn_mean, bn_var, Y);

  // 4: 1x1 conv + bias -> out [1,10,32,32] (exact grid)
  k_conv1x1<<<(NCLS * SPAT) / TB, TB, 0, stream>>>(Y, conv2_w, conv2_b, out);
}
